// NODAE_13915694039428
// MI455X (gfx1250) — matrix operand order repaired and checked by reading, not compiled, not run on hardware
//
#include <hip/hip_runtime.h>
#include <hip/hip_bf16.h>

#define BATCH  262144
#define SDIM   17
#define ADIM   6
#define HDIM   20
#define LATD   2
#define CHD    5
#define KW     5
#define NSTEPS 40
#define NTILES 21

typedef __attribute__((ext_vector_type(16))) _Float16 v16h;
typedef __attribute__((ext_vector_type(2)))  __fp16   v2fp;
typedef __attribute__((ext_vector_type(8)))  float    v8f;

// weight tile ids (order shared by k0_pack and k1_main)
#define T_ENC1_0  0
#define T_ENC1_1  1
#define T_ENC2_0  2
#define T_OO1_0   3
#define T_OO1_1   4
#define T_OO2_0   5
#define T_OO2_1   6
#define T_OO3_0   7
#define T_DEC1_0  8
#define T_DEC2_0  9
#define T_DEC2_1  10
#define T_RI_0    11
#define T_RI_1    12
#define T_OR1_0   13
#define T_OR1_1   14
#define T_OR2_0   15
#define T_OR2_1   16
#define T_OR3_0   17
#define T_OR3_1   18
#define T_EF_0    19
#define T_EF_1    20

// ---------------------------------------------------------------------------
// helpers
// ---------------------------------------------------------------------------
__device__ __forceinline__ v8f vzero() {
    v8f z = {0.f,0.f,0.f,0.f,0.f,0.f,0.f,0.f};
    return z;
}

__device__ __forceinline__ v8f vmad(v8f a, float s, v8f b) {
    v8f r;
#pragma unroll
    for (int i = 0; i < 8; ++i) r[i] = a[i] + s * b[i];
    return r;
}

__device__ __forceinline__ v8f elu8(v8f x) {
    v8f r;
#pragma unroll
    for (int i = 0; i < 8; ++i) {
        float v = x[i];
        r[i] = v > 0.f ? v : (__expf(v) - 1.f);
    }
    return r;
}

// single v_cvt_pk_rtz_f16_f32
__device__ __forceinline__ int pk2(float a, float b) {
    union { v2fp h; int i; } u;
    u.h = __builtin_amdgcn_cvt_pkrtz(a, b);
    return u.i;
}

// lane <-> lane^16 exchange: ds_swizzle group-of-32, xor=0x10, and=0x1f
__device__ __forceinline__ int swap16(int x) {
    return __builtin_amdgcn_ds_swizzle(x, 0x401F);
}

// D = A(16x32 f16) x B(32x16 f16) + C(f32)
__device__ __forceinline__ v8f wmma16(v16h a, v16h b, v8f c) {
    return __builtin_amdgcn_wmma_f32_16x16x32_f16(
        /*neg_a=*/false, a, /*neg_b=*/false, b,
        /*c_mod=*/(short)0, c, /*reuse_a=*/false, /*reuse_b=*/false);
}

// Fetch pre-packed A-operand tile: [tile][lane][16 halfs] -> 32B/lane
__device__ __forceinline__ v16h load_tile(const _Float16* __restrict__ wt, int t) {
    const int lane = threadIdx.x & 31;
    return *(const v16h*)(wt + t * 512 + lane * 16);
}

// Build B operand (32x16 f16; K = feature, N = batch column) from a state held
// in D format (d0 = features 0..15, d1 = features 16..31).
// B layout used here (contiguous k, same as k0_pack's A): lane n holds K=0..15 of column n; lane n+16 holds K=16..31.
__device__ __forceinline__ v16h make_B(v8f d0, v8f d1) {
    const bool lo = (threadIdx.x & 16) == 0;
    union { int i[8]; v16h v; } u;
#pragma unroll
    for (int j = 0; j < 4; ++j) {
        const int p0 = pk2(d0[2 * j], d0[2 * j + 1]);
        const int p1 = pk2(d1[2 * j], d1[2 * j + 1]);
        const int q0 = swap16(p0);
        const int q1 = swap16(p1);
        u.i[j]     = lo ? p0 : q1;   // K = 0..7   : 16..23
        u.i[4 + j] = lo ? q0 : p1;   // K = 8..15  : 24..31
    }
    return u.v;
}

// Same, but plant the constant-1 "bias feature" at K == KK before packing.
// (The aliased slot multiplies a zero-padded weight column, so it is safe.)
template <int KK>
__device__ __forceinline__ v16h make_Bb(v8f d0, v8f d1) {
    if constexpr (KK < 8)        d0[KK]      = 1.0f;
    else if constexpr (KK < 16)  d0[KK - 8]  = 1.0f;
    else if constexpr (KK < 24)  d1[KK - 16] = 1.0f;
    else                         d1[KK - 24] = 1.0f;
    return make_B(d0, d1);
}

struct Tile2 { v16h a0, a1; };

// odefunc, 20-dim state: elu(W1 y) -> elu(W2 h) -> W3 h   (biases folded)
__device__ __forceinline__ void ode20(v8f y0, v8f y1,
                                      const Tile2& w1, const Tile2& w2,
                                      const Tile2& w3, v8f& o0, v8f& o1) {
    const v8f z = vzero();
    v16h b = make_Bb<HDIM>(y0, y1);
    v8f h0 = elu8(wmma16(w1.a0, b, z));
    v8f h1 = elu8(wmma16(w1.a1, b, z));
    b = make_Bb<HDIM>(h0, h1);
    h0 = elu8(wmma16(w2.a0, b, z));
    h1 = elu8(wmma16(w2.a1, b, z));
    b = make_Bb<HDIM>(h0, h1);
    o0 = wmma16(w3.a0, b, z);
    o1 = wmma16(w3.a1, b, z);
}

// odefunc, 2-dim latent state (W1: 20x2, W2: 20x20, W3: 2x20 -> single tile)
__device__ __forceinline__ v8f ode2f(v8f y, const Tile2& w1, const Tile2& w2,
                                     v16h w3) {
    const v8f z = vzero();
    v16h b = make_Bb<LATD>(y, vzero());
    v8f h0 = elu8(wmma16(w1.a0, b, z));
    v8f h1 = elu8(wmma16(w1.a1, b, z));
    b = make_Bb<HDIM>(h0, h1);
    h0 = elu8(wmma16(w2.a0, b, z));
    h1 = elu8(wmma16(w2.a1, b, z));
    b = make_Bb<HDIM>(h0, h1);
    return wmma16(w3, b, z);
}

// ---------------------------------------------------------------------------
// Kernel 0: pack every weight matrix (+bias folded at column K=kk) into the
// 16x32 f16 A-operand lane image: wt[tile][lane][16 halfs]. Runs once per
// launch; removes all guarded scalar weight loads from the hot kernel.
// ---------------------------------------------------------------------------
struct WDesc { const float* W; const float* b; int nr, kk, t; };

__global__ __launch_bounds__(256) void k0_pack(
    const float* enc_w1, const float* enc_b1,
    const float* enc_w2, const float* enc_b2,
    const float* oo_w1,  const float* oo_b1,
    const float* oo_w2,  const float* oo_b2,
    const float* oo_w3,  const float* oo_b3,
    const float* dec_w1, const float* dec_b1,
    const float* dec_w2, const float* dec_b2,
    const float* ri_w,   const float* ri_b,
    const float* or_w1,  const float* or_b1,
    const float* or_w2,  const float* or_b2,
    const float* or_w3,  const float* or_b3,
    const float* eb_fc_w,const float* eb_fc_b,
    _Float16* __restrict__ wt)
{
    const WDesc d[NTILES] = {
        {enc_w1, enc_b1, SDIM, SDIM, 0}, {enc_w1, enc_b1, SDIM, SDIM, 1},
        {enc_w2, enc_b2, LATD, SDIM, 0},
        {oo_w1,  oo_b1,  HDIM, LATD, 0}, {oo_w1,  oo_b1,  HDIM, LATD, 1},
        {oo_w2,  oo_b2,  HDIM, HDIM, 0}, {oo_w2,  oo_b2,  HDIM, HDIM, 1},
        {oo_w3,  oo_b3,  LATD, HDIM, 0},
        {dec_w1, dec_b1, LATD, LATD, 0},
        {dec_w2, dec_b2, SDIM, LATD, 0}, {dec_w2, dec_b2, SDIM, LATD, 1},
        {ri_w,   ri_b,   HDIM, SDIM + ADIM, 0},
        {ri_w,   ri_b,   HDIM, SDIM + ADIM, 1},
        {or_w1,  or_b1,  HDIM, HDIM, 0}, {or_w1,  or_b1,  HDIM, HDIM, 1},
        {or_w2,  or_b2,  HDIM, HDIM, 0}, {or_w2,  or_b2,  HDIM, HDIM, 1},
        {or_w3,  or_b3,  HDIM, HDIM, 0}, {or_w3,  or_b3,  HDIM, HDIM, 1},
        {eb_fc_w, eb_fc_b, HDIM, HDIM, 0}, {eb_fc_w, eb_fc_b, HDIM, HDIM, 1}
    };
    const int t = blockIdx.x;
    const WDesc dd = d[t];
    for (int e = threadIdx.x; e < 512; e += 256) {
        const int lane = e >> 4;
        const int hidx = e & 15;
        const int vv = hidx >> 1, hh = hidx & 1;
        int kb = 2 * vv;
        if (lane >= 16) kb += 16;
        const int k   = kb + hh;
        const int row = dd.t * 16 + (lane & 15);
        float val = 0.0f;
        if (row < dd.nr) {
            if (k < dd.kk)       val = dd.W[row * dd.kk + k];
            else if (k == dd.kk) val = dd.b[row];
        }
        wt[t * 512 + e] = (_Float16)val;
    }
}

// ---------------------------------------------------------------------------
// Kernel 1: encoder -> obs ODE -> decoder (out_obs) ; ri -> reward ODE ->
// eb_fc (-> ws_r2). One wave handles 16 batch rows. 8 waves / block.
// ---------------------------------------------------------------------------
__global__ __launch_bounds__(256) void k1_main(
    const float* __restrict__ x,
    const _Float16* __restrict__ wt,
    float* __restrict__ out_obs, float* __restrict__ ws_r2)
{
    const int  lane = threadIdx.x & 31;
    const int  wave = threadIdx.x >> 5;
    const long tile = (long)blockIdx.x * 8 + wave;
    const long base = tile * 16;
    const int  c    = lane & 15;
    const long rowi = base + c;

    const float hs = 1.0f / (float)NSTEPS;
    const v8f z = vzero();

    // ---- stage the wave's 16 input rows (368 contiguous floats) in LDS ----
    __shared__ float xs[8][376];
    float* xw = xs[wave];
    {
        const float* xg = x + base * (SDIM + ADIM);
#pragma unroll
        for (int i = 0; i < 12; ++i) {
            const int idx = lane + 32 * i;
            if (idx < 16 * (SDIM + ADIM)) xw[idx] = xg[idx];
        }
    }

    // ---- build B operands for encoder input s = x[:,:17] and ri input x ----
    union { int i[8]; v16h v; } bsU, brU;
    {
        const float* xr = xw + c * (SDIM + ADIM);
        if (lane < 16) {        // K = 0..15: shared by both
#pragma unroll
            for (int j = 0; j < 8; ++j) bsU.i[j] = pk2(xr[2 * j], xr[2 * j + 1]);
#pragma unroll
            for (int j = 0; j < 8; ++j) brU.i[j] = bsU.i[j];
        } else {                // K = 16..31
            bsU.i[0] = pk2(xr[16], 1.0f);           // s[16], bias@K=17
#pragma unroll
            for (int j = 1; j < 8; ++j) bsU.i[j] = 0;
            brU.i[0] = pk2(xr[16], xr[17]);
            brU.i[1] = pk2(xr[18], xr[19]);
            brU.i[2] = pk2(xr[20], xr[21]);
            brU.i[3] = pk2(xr[22], 1.0f);           // bias@K=23
            brU.i[4] = brU.i[5] = brU.i[6] = brU.i[7] = 0;
        }
    }

    // ---- encoder: lat = (s W1^T + b1) W2^T + b2 (no nonlinearity) ----
    v8f lat0;
    {
        Tile2 ew1 = { load_tile(wt, T_ENC1_0), load_tile(wt, T_ENC1_1) };
        v8f h0 = wmma16(ew1.a0, bsU.v, z);
        v8f h1 = wmma16(ew1.a1, bsU.v, z);
        v16h bh = make_Bb<SDIM>(h0, h1);
        lat0 = wmma16(load_tile(wt, T_ENC2_0), bh, z);
    }

    // ---- obs ODE: RK4 x 40 on 2-dim latent ----
    {
        Tile2 w1 = { load_tile(wt, T_OO1_0), load_tile(wt, T_OO1_1) };
        Tile2 w2 = { load_tile(wt, T_OO2_0), load_tile(wt, T_OO2_1) };
        v16h w3  =   load_tile(wt, T_OO3_0);
#pragma unroll 1
        for (int s = 0; s < NSTEPS; ++s) {
            v8f k1 = ode2f(lat0, w1, w2, w3);
            v8f k2 = ode2f(vmad(lat0, 0.5f * hs, k1), w1, w2, w3);
            v8f k3 = ode2f(vmad(lat0, 0.5f * hs, k2), w1, w2, w3);
            v8f k4 = ode2f(vmad(lat0, hs, k3), w1, w2, w3);
#pragma unroll
            for (int i = 0; i < 8; ++i)
                lat0[i] += (hs / 6.0f) * (k1[i] + 2.0f * (k2[i] + k3[i]) + k4[i]);
        }
    }

    // ---- decoder -> out_obs ----
    {
        v16h bl  = make_Bb<LATD>(lat0, vzero());
        v8f d    = wmma16(load_tile(wt, T_DEC1_0), bl, z);
        v16h bd  = make_Bb<LATD>(d, vzero());
        v8f o0 = wmma16(load_tile(wt, T_DEC2_0), bd, z);
        v8f o1 = wmma16(load_tile(wt, T_DEC2_1), bd, z);
        float* op = out_obs + rowi * SDIM;
        if (lane < 16) {
#pragma unroll
            for (int r = 0; r < 8; ++r) op[r] = o0[r];
            op[16] = o1[0];
        } else {
#pragma unroll
            for (int r = 0; r < 8; ++r) op[8 + r] = o0[r];
        }
    }

    // ---- reward path: r = ri(x); RK4 x 40; eb_fc; write to ws ----
    v8f r0, r1;
    {
        r0 = wmma16(load_tile(wt, T_RI_0), brU.v, z);
        r1 = wmma16(load_tile(wt, T_RI_1), brU.v, z);
    }
    {
        Tile2 w1 = { load_tile(wt, T_OR1_0), load_tile(wt, T_OR1_1) };
        Tile2 w2 = { load_tile(wt, T_OR2_0), load_tile(wt, T_OR2_1) };
        Tile2 w3 = { load_tile(wt, T_OR3_0), load_tile(wt, T_OR3_1) };
#pragma unroll 1
        for (int s = 0; s < NSTEPS; ++s) {
            v8f k10, k11, k20, k21, k30, k31, k40, k41;
            ode20(r0, r1, w1, w2, w3, k10, k11);
            ode20(vmad(r0, 0.5f * hs, k10), vmad(r1, 0.5f * hs, k11),
                  w1, w2, w3, k20, k21);
            ode20(vmad(r0, 0.5f * hs, k20), vmad(r1, 0.5f * hs, k21),
                  w1, w2, w3, k30, k31);
            ode20(vmad(r0, hs, k30), vmad(r1, hs, k31), w1, w2, w3, k40, k41);
#pragma unroll
            for (int i = 0; i < 8; ++i) {
                r0[i] += (hs / 6.0f) * (k10[i] + 2.0f * (k20[i] + k30[i]) + k40[i]);
                r1[i] += (hs / 6.0f) * (k11[i] + 2.0f * (k21[i] + k31[i]) + k41[i]);
            }
        }
    }
    {
        v16h br2 = make_Bb<HDIM>(r0, r1);
        v8f t0 = wmma16(load_tile(wt, T_EF_0), br2, z);
        v8f t1 = wmma16(load_tile(wt, T_EF_1), br2, z);
        float* wp = ws_r2 + rowi * HDIM;
        if (lane < 16) {
#pragma unroll
            for (int r = 0; r < 8; ++r) wp[r] = t0[r];
#pragma unroll
            for (int r = 0; r < 4; ++r) wp[16 + r] = t1[r];
        } else {
#pragma unroll
            for (int r = 0; r < 8; ++r) wp[8 + r] = t0[r];
        }
    }
}

// ---------------------------------------------------------------------------
// Kernel 2: eb conv per row + per-channel sum / sumsq (global BN stats)
// ---------------------------------------------------------------------------
__global__ __launch_bounds__(256) void k2_stats(
    const float* __restrict__ r2, const float* __restrict__ cw,
    const float* __restrict__ cb, float* __restrict__ acc)
{
    const long b = (long)blockIdx.x * 256 + threadIdx.x;
    float row[HDIM + 4];
    row[0] = row[1] = row[HDIM + 2] = row[HDIM + 3] = 0.f;
    const float* rp = r2 + b * HDIM;
#pragma unroll
    for (int i = 0; i < HDIM; ++i) row[2 + i] = rp[i];

    float s[CHD], q[CHD];
#pragma unroll
    for (int ch = 0; ch < CHD; ++ch) {
        float su = 0.f, sq = 0.f;
        const float bb = cb[ch];
#pragma unroll
        for (int p = 0; p < HDIM; ++p) {
            float v = bb;
#pragma unroll
            for (int j = 0; j < KW; ++j) v += cw[ch * KW + j] * row[p + j];
            su += v; sq += v * v;
        }
        s[ch] = su; q[ch] = sq;
    }
#pragma unroll
    for (int off = 16; off > 0; off >>= 1) {
#pragma unroll
        for (int ch = 0; ch < CHD; ++ch) {
            s[ch] += __shfl_down(s[ch], off, 32);
            q[ch] += __shfl_down(q[ch], off, 32);
        }
    }
    __shared__ float lacc[2 * CHD];
    if (threadIdx.x < 2 * CHD) lacc[threadIdx.x] = 0.f;
    __syncthreads();
    if ((threadIdx.x & 31) == 0) {
#pragma unroll
        for (int ch = 0; ch < CHD; ++ch) {
            atomicAdd(&lacc[ch], s[ch]);
            atomicAdd(&lacc[CHD + ch], q[ch]);
        }
    }
    __syncthreads();
    if (threadIdx.x < 2 * CHD) atomicAdd(&acc[threadIdx.x], lacc[threadIdx.x]);
}

// ---------------------------------------------------------------------------
// Kernel 3: finalize BN scale/shift
// ---------------------------------------------------------------------------
__global__ void k3_finalize(const float* __restrict__ acc,
                            const float* __restrict__ g,
                            const float* __restrict__ bt,
                            float* __restrict__ prm)
{
    const int i = threadIdx.x;
    if (i < CHD) {
        const float n = (float)BATCH * (float)HDIM;
        const float mu  = acc[i] / n;
        const float var = acc[CHD + i] / n - mu * mu;
        const float sc  = g[i] * rsqrtf(var + 1e-5f);
        prm[i] = sc;
        prm[CHD + i] = bt[i] - mu * sc;
    }
}

// ---------------------------------------------------------------------------
// Kernel 4: BN+relu, two conv/pool/fc blocks, channel max, reward head.
// One thread per batch row (cheap VALU tail, ~2% of total FLOPs).
// ---------------------------------------------------------------------------
__global__ __launch_bounds__(256) void k4_tail(
    const float* __restrict__ r2,  const float* __restrict__ prm,
    const float* __restrict__ ecw, const float* __restrict__ ecb,
    const float* __restrict__ c1w, const float* __restrict__ c1b,
    const float* __restrict__ f1w, const float* __restrict__ f1b,
    const float* __restrict__ c2w, const float* __restrict__ c2b,
    const float* __restrict__ f2w, const float* __restrict__ f2b,
    const float* __restrict__ row_, const float* __restrict__ rob,
    float* __restrict__ out_rew)
{
    const long b = (long)blockIdx.x * 256 + threadIdx.x;
    float row[HDIM + 4];
    row[0] = row[1] = row[HDIM + 2] = row[HDIM + 3] = 0.f;
    const float* rp = r2 + b * HDIM;
#pragma unroll
    for (int i = 0; i < HDIM; ++i) row[2 + i] = rp[i];

    float c0[CHD][HDIM + 4];
#pragma unroll
    for (int ch = 0; ch < CHD; ++ch) {
        c0[ch][0] = c0[ch][1] = c0[ch][HDIM + 2] = c0[ch][HDIM + 3] = 0.f;
        const float sc = prm[ch], sh = prm[CHD + ch];
#pragma unroll
        for (int p = 0; p < HDIM; ++p) {
            float v = ecb[ch];
#pragma unroll
            for (int j = 0; j < KW; ++j) v += ecw[ch * KW + j] * row[p + j];
            v = v * sc + sh;
            c0[ch][2 + p] = v > 0.f ? v : 0.f;
        }
    }

    const float* cw = c1w; const float* cbv = c1b;
    const float* fw = f1w; const float* fb  = f1b;
#pragma unroll
    for (int blk = 0; blk < 2; ++blk) {
        float m[CHD][HDIM / 2];
#pragma unroll
        for (int ch = 0; ch < CHD; ++ch) {
            float t[HDIM];
#pragma unroll
            for (int p = 0; p < HDIM; ++p) {
                float v = cbv[ch];
#pragma unroll
                for (int ic = 0; ic < CHD; ++ic)
#pragma unroll
                    for (int j = 0; j < KW; ++j)
                        v += cw[(ch * CHD + ic) * KW + j] * c0[ic][p + j];
                t[p] = v;
            }
#pragma unroll
            for (int qq = 0; qq < HDIM / 2; ++qq)
                m[ch][qq] = fmaxf(t[2 * qq], t[2 * qq + 1]);
        }
#pragma unroll
        for (int ch = 0; ch < CHD; ++ch) {
#pragma unroll
            for (int n = 0; n < HDIM; ++n) {
                float v = fb[n];
#pragma unroll
                for (int qq = 0; qq < HDIM / 2; ++qq)
                    v += m[ch][qq] * fw[n * (HDIM / 2) + qq];
                c0[ch][2 + n] = v > 0.f ? v : 0.f;
            }
        }
        cw = c2w; cbv = c2b; fw = f2w; fb = f2b;
    }

    float o = rob[0];
#pragma unroll
    for (int n = 0; n < HDIM; ++n) {
        float rm = c0[0][2 + n];
#pragma unroll
        for (int ch = 1; ch < CHD; ++ch) rm = fmaxf(rm, c0[ch][2 + n]);
        o += rm * row_[n];
    }
    out_rew[b] = o;
}

// ---------------------------------------------------------------------------
extern "C" void kernel_launch(void* const* d_in, const int* in_sizes, int n_in,
                              void* d_out, int out_size, void* d_ws, size_t ws_size,
                              hipStream_t stream) {
    const float* x      = (const float*)d_in[0];
    const float* enc_w1 = (const float*)d_in[1];
    const float* enc_b1 = (const float*)d_in[2];
    const float* enc_w2 = (const float*)d_in[3];
    const float* enc_b2 = (const float*)d_in[4];
    const float* oo_w1  = (const float*)d_in[5];
    const float* oo_b1  = (const float*)d_in[6];
    const float* oo_w2  = (const float*)d_in[7];
    const float* oo_b2  = (const float*)d_in[8];
    const float* oo_w3  = (const float*)d_in[9];
    const float* oo_b3  = (const float*)d_in[10];
    const float* dec_w1 = (const float*)d_in[11];
    const float* dec_b1 = (const float*)d_in[12];
    const float* dec_w2 = (const float*)d_in[13];
    const float* dec_b2 = (const float*)d_in[14];
    const float* ri_w   = (const float*)d_in[15];
    const float* ri_b   = (const float*)d_in[16];
    const float* or_w1  = (const float*)d_in[17];
    const float* or_b1  = (const float*)d_in[18];
    const float* or_w2  = (const float*)d_in[19];
    const float* or_b2  = (const float*)d_in[20];
    const float* or_w3  = (const float*)d_in[21];
    const float* or_b3  = (const float*)d_in[22];
    const float* ebfw   = (const float*)d_in[23];
    const float* ebfb   = (const float*)d_in[24];
    const float* ecw    = (const float*)d_in[25];
    const float* ecb    = (const float*)d_in[26];
    const float* bng    = (const float*)d_in[27];
    const float* bnb    = (const float*)d_in[28];
    const float* b1cw   = (const float*)d_in[29];
    const float* b1cb   = (const float*)d_in[30];
    const float* b1fw   = (const float*)d_in[31];
    const float* b1fb   = (const float*)d_in[32];
    const float* b2cw   = (const float*)d_in[33];
    const float* b2cb   = (const float*)d_in[34];
    const float* b2fw   = (const float*)d_in[35];
    const float* b2fb   = (const float*)d_in[36];
    const float* ro_w   = (const float*)d_in[37];
    const float* ro_b   = (const float*)d_in[38];

    // workspace layout
    float*    ws_r2 = (float*)d_ws;                                 // B*20 f32
    _Float16* wt    = (_Float16*)((char*)d_ws + (size_t)BATCH * HDIM * 4);
    float*    acc   = (float*)((char*)d_ws + (size_t)BATCH * HDIM * 4
                               + (size_t)NTILES * 512 * 2);         // 10 f32
    float*    prm   = acc + 16;                                     // 10 f32

    float* out_obs = (float*)d_out;
    float* out_rew = out_obs + (size_t)BATCH * SDIM;

    (void)hipMemsetAsync(acc, 0, 2 * CHD * sizeof(float), stream);

    k0_pack<<<NTILES, 256, 0, stream>>>(
        enc_w1, enc_b1, enc_w2, enc_b2,
        oo_w1, oo_b1, oo_w2, oo_b2, oo_w3, oo_b3,
        dec_w1, dec_b1, dec_w2, dec_b2,
        ri_w, ri_b, or_w1, or_b1, or_w2, or_b2, or_w3, or_b3,
        ebfw, ebfb, wt);

    k1_main<<<BATCH / 128, 256, 0, stream>>>(x, wt, out_obs, ws_r2);

    k2_stats<<<BATCH / 256, 256, 0, stream>>>(ws_r2, ecw, ecb, acc);
    k3_finalize<<<1, 32, 0, stream>>>(acc, bng, bnb, prm);
    k4_tail<<<BATCH / 256, 256, 0, stream>>>(
        ws_r2, prm, ecw, ecb, b1cw, b1cb, b1fw, b1fb,
        b2cw, b2cb, b2fw, b2fb, ro_w, ro_b, out_rew);
}
